// MHA_10703058502098
// MI455X (gfx1250) — matrix operand order repaired and checked by reading, compiled, not run on hardware
//
#include <hip/hip_runtime.h>

// MHA forward, MI455X (gfx1250, wave32).
// Stage 1: QKV GEMM  (f16 WMMA 16x16x32, fp32 accum) -> f16 Q/K/V workspaces
// Stage 2: flash attention (S^T formulation), online softmax fp32,
//          K/V tiles double-buffered via GLOBAL_LOAD_ASYNC_TO_LDS_B128
//          (ASYNCcnt, partial wait s_wait_asynccnt 0x4 for overlap)
// Stage 3: output projection GEMM -> fp32 d_out

#define DIMD   768
#define NHEADS 12
#define DHEAD  64
#define SEQ    2048
#define BATCH  2
#define ROWS   (BATCH * SEQ)              // 4096
#define QKVCOL (3 * DIMD)                 // 2304
#define SCALEF 0.03608439182435161f       // 768^-0.5  (reference uses dim, not dhead)

typedef _Float16 half_t;
typedef __attribute__((ext_vector_type(16))) _Float16 v16h;
typedef __attribute__((ext_vector_type(8)))  float    v8f;

// ---- WMMA fragment index maps (wave32; A/B use contiguous k, not ISA table order) -------------------
// A (16x32 f16, MxK): row = lane&15, VGPR j -> K = 16*hi + 2*j
__device__ __forceinline__ int kA_of(int j, int hi) {
  return hi * 16 + 2 * j;
}
// B (32x16 f16, KxN): col = lane&15, VGPR j -> K = 16*hi + 2*j
__device__ __forceinline__ int kB_of(int j, int hi) { return hi * 16 + 2 * j; }
// C/D (16x16 f32): element r -> M = r + 8*hi, N = lane&15

// p points at a K-contiguous row (stride 1 in K); pairs are 4B-aligned.
__device__ __forceinline__ v16h load_fragA(const half_t* p, int hi) {
  union { unsigned u[8]; v16h v; } t;
#pragma unroll
  for (int j = 0; j < 8; ++j) t.u[j] = *(const unsigned*)(p + kA_of(j, hi));
  return t.v;
}
__device__ __forceinline__ v16h load_fragB(const half_t* p, int hi) {
  union { unsigned u[8]; v16h v; } t;
#pragma unroll
  for (int j = 0; j < 8; ++j) t.u[j] = *(const unsigned*)(p + kB_of(j, hi));
  return t.v;
}

__device__ __forceinline__ v8f wmma_f16(v16h a, v16h b, v8f c) {
  return __builtin_amdgcn_wmma_f32_16x16x32_f16(false, a, false, b, (short)0, c,
                                                false, false);
}

// Async 16B copy global -> LDS, tracked by ASYNCcnt (CDNA5 §15.18.3 op 98).
// LDS offset = low 32 bits of the generic address (LDS aperture, ISA 10.2).
__device__ __forceinline__ void async_copy_b128(void* lds_ptr, const void* gptr) {
  unsigned lds_off = (unsigned)(size_t)lds_ptr;
  unsigned long long ga = (unsigned long long)(size_t)gptr;
  asm volatile("global_load_async_to_lds_b128 %0, %1, off"
               :: "v"(lds_off), "v"(ga)
               : "memory");
}

// ---------------------------------------------------------------------------
// GEMM: C[64x128 tile] = A[rows x 768] * W[768 x ldw] (+bias). 8 waves/block,
// each wave computes a 16-col strip (4 row sub-tiles -> 4 WMMAs per K-step).
// MODE 0: scatter to f16 Q[b,h,n,64], K[b,h,n,64], V^T[b,h,64,n]
// MODE 1: fp32 out[row, col] = acc + bias
// ---------------------------------------------------------------------------
template <int MODE>
__global__ void __launch_bounds__(256) gemm_wmma(
    const float* __restrict__ A, const float* __restrict__ W,
    const float* __restrict__ bias, int ldw,
    half_t* __restrict__ Qh, half_t* __restrict__ Kh, half_t* __restrict__ Vt,
    float* __restrict__ Out) {
  __shared__ __align__(16) half_t Xs[64 * 32];    // X tile, row-major (K contig)
  __shared__ __align__(16) half_t Wt[128 * 32];   // W tile, transposed (K contig)

  const int tid  = threadIdx.x;
  const int wave = tid >> 5;
  const int lane = tid & 31;
  const int ln15 = lane & 15;
  const int hi   = lane >> 4;
  const int rowBase = blockIdx.y * 64;
  const int colBase = blockIdx.x * 128;

  v8f acc[4];
#pragma unroll
  for (int i = 0; i < 4; ++i)
#pragma unroll
    for (int r = 0; r < 8; ++r) acc[i][r] = 0.f;

  const int xr = tid >> 2, xseg = (tid & 3) * 8;       // X staging map
  const int wk = tid >> 3, wcseg = (tid & 7) * 16;     // W staging map

  for (int k0 = 0; k0 < DIMD; k0 += 32) {
    __syncthreads();
    {  // stage X: 64x32 -> f16, 8 elems/thread, coalesced along K
      const float* src = A + (size_t)(rowBase + xr) * DIMD + k0 + xseg;
      half_t* dst = Xs + xr * 32 + xseg;
#pragma unroll
      for (int i = 0; i < 8; ++i) dst[i] = (half_t)src[i];
      if (k0 + 32 < DIMD) __builtin_prefetch(src + 32, 0, 1);
    }
    {  // stage W transposed: 32x128 -> Wt[col][k], 16 elems/thread
      const float* src = W + (size_t)(k0 + wk) * ldw + colBase + wcseg;
#pragma unroll
      for (int i = 0; i < 16; ++i) Wt[(wcseg + i) * 32 + wk] = (half_t)src[i];
      if (k0 + 32 < DIMD) __builtin_prefetch(src + (size_t)32 * ldw, 0, 1);
    }
    __syncthreads();

    v16h bfrag = load_fragB(Wt + (wave * 16 + ln15) * 32, hi);
#pragma unroll
    for (int rs = 0; rs < 4; ++rs) {
      v16h afrag = load_fragA(Xs + (rs * 16 + ln15) * 32, hi);
      acc[rs] = wmma_f16(afrag, bfrag, acc[rs]);
    }
  }

  const int cg = colBase + wave * 16 + ln15;
  const float bv = bias[cg];
  if (MODE == 0) {
    const int s   = cg / DIMD;           // 0=q 1=k 2=v
    const int rem = cg - s * DIMD;
    const int hh  = rem >> 6;
    const int dc  = rem & 63;
#pragma unroll
    for (int rs = 0; rs < 4; ++rs) {
#pragma unroll
      for (int r = 0; r < 8; ++r) {
        int rg = rowBase + rs * 16 + r + 8 * hi;     // M = r + 8*hi
        int bi = rg >> 11, ni = rg & 2047;
        half_t hv = (half_t)(acc[rs][r] + bv);
        if (s == 0)
          Qh[(((size_t)bi * NHEADS + hh) * SEQ + ni) * DHEAD + dc] = hv;
        else if (s == 1)
          Kh[(((size_t)bi * NHEADS + hh) * SEQ + ni) * DHEAD + dc] = hv;
        else
          Vt[(((size_t)bi * NHEADS + hh) * DHEAD + dc) * SEQ + ni] = hv;
      }
    }
  } else {
#pragma unroll
    for (int rs = 0; rs < 4; ++rs) {
#pragma unroll
      for (int r = 0; r < 8; ++r) {
        int rg = rowBase + rs * 16 + r + 8 * hi;
        Out[(size_t)rg * DIMD + cg] = acc[rs][r] + bv;
      }
    }
  }
}

// ---------------------------------------------------------------------------
// Flash attention. Block = 4 waves = 64 queries of one (b,h); waves share
// double-buffered K/V LDS tiles of 32 keys filled by async global->LDS
// copies (tile kt+1 in flight behind tile kt's 8 WMMAs; async loads complete
// in order, so s_wait_asynccnt 0x4 releases tile kt while kt+1 stays
// outstanding). fp32 online softmax; P^T bounced via per-wave LDS.
// ---------------------------------------------------------------------------
__global__ void __launch_bounds__(128) attn_kernel(
    const half_t* __restrict__ Qh, const half_t* __restrict__ Kh,
    const half_t* __restrict__ Vt, float* __restrict__ O) {
  __shared__ __align__(16) half_t Ks[2][32 * 64];   // [buf][key][d]
  __shared__ __align__(16) half_t Vs[2][64 * 32];   // [buf][d][key]  (V^T)
  __shared__ __align__(16) half_t Ps[4][16 * 32];   // per-wave [q][key]

  const int tid  = threadIdx.x;
  const int wave = tid >> 5;
  const int lane = tid & 31;
  const int ln15 = lane & 15;
  const int hi   = lane >> 4;
  const int bh   = blockIdx.y;                      // 0..23
  const int bi   = bh / NHEADS;
  const int h    = bh - bi * NHEADS;
  const int qg   = blockIdx.x * 64 + wave * 16 + ln15;

  // Q^T B-fragments (d = contraction): b[2j] = Q[q][16*hi + 2j (+32*dk)]
  v16h bq[2];
  const half_t* qptr = Qh + ((size_t)bh * SEQ + qg) * DHEAD + hi * 16;
#pragma unroll
  for (int dk = 0; dk < 2; ++dk) {
    union { uint4 u[2]; v16h v; } t;
    const uint4* p = (const uint4*)(qptr + dk * 32);
    t.u[0] = p[0]; t.u[1] = p[1];
    bq[dk] = t.v;
  }

  v8f acc[4];
#pragma unroll
  for (int i = 0; i < 4; ++i)
#pragma unroll
    for (int r = 0; r < 8; ++r) acc[i][r] = 0.f;
  float m = -1e30f, l = 0.f;

  // staging maps (per-thread, loop-invariant)
  const int kkey = tid >> 2, kseg = (tid & 3) * 16;  // K: 32x64 halfs
  const int vd   = tid >> 1, vseg = (tid & 1) * 16;  // V^T: 64x32 halfs
  const half_t* Kg = Kh + ((size_t)bh * SEQ + kkey) * DHEAD + kseg;
  const half_t* Vg = Vt + ((size_t)bh * DHEAD + vd) * SEQ + vseg;
  half_t* KsDst[2] = {Ks[0] + kkey * 64 + kseg, Ks[1] + kkey * 64 + kseg};
  half_t* VsDst[2] = {Vs[0] + vd * 32 + vseg,   Vs[1] + vd * 32 + vseg};

  // prologue: tile 0 -> buffer 0
  async_copy_b128(KsDst[0],     Kg);
  async_copy_b128(KsDst[0] + 8, Kg + 8);
  async_copy_b128(VsDst[0],     Vg);
  async_copy_b128(VsDst[0] + 8, Vg + 8);
  Kg += 32 * DHEAD;
  Vg += 32;

  for (int kt = 0; kt < SEQ / 32; ++kt) {
    const int p = kt & 1;
    __syncthreads();   // all waves done reading buf[1-p] (iter kt-1)
    if (kt + 1 < SEQ / 32) {
      async_copy_b128(KsDst[1 - p],     Kg);
      async_copy_b128(KsDst[1 - p] + 8, Kg + 8);
      async_copy_b128(VsDst[1 - p],     Vg);
      async_copy_b128(VsDst[1 - p] + 8, Vg + 8);
      Kg += 32 * DHEAD;
      Vg += 32;
      // in-order completion: <=4 outstanding => tile kt's 4 copies landed
      asm volatile("s_wait_asynccnt 0x4" ::: "memory");
    } else {
      asm volatile("s_wait_asynccnt 0x0" ::: "memory");
    }
    __syncthreads();   // tile kt visible to all waves
    const half_t* Kb = Ks[p];
    const half_t* Vb = Vs[p];

    // S^T = K_tile x Q^T : two 16x16 tiles (keys 0-15, 16-31), d in 2 chunks
    v8f s0, s1;
#pragma unroll
    for (int r = 0; r < 8; ++r) { s0[r] = 0.f; s1[r] = 0.f; }
#pragma unroll
    for (int dk = 0; dk < 2; ++dk) {
      v16h ka0 = load_fragA(Kb + (ln15)      * 64 + dk * 32, hi);
      v16h ka1 = load_fragA(Kb + (16 + ln15) * 64 + dk * 32, hi);
      s0 = wmma_f16(ka0, bq[dk], s0);
      s1 = wmma_f16(ka1, bq[dk], s1);
    }

    // online softmax over keys (per query column q = ln15)
    float tmax = -1e30f;
#pragma unroll
    for (int r = 0; r < 8; ++r) {
      s0[r] *= SCALEF; s1[r] *= SCALEF;
      tmax = fmaxf(tmax, fmaxf(s0[r], s1[r]));
    }
    tmax = fmaxf(tmax, __shfl_xor(tmax, 16));
    float mn   = fmaxf(m, tmax);
    float corr = __expf(m - mn);
    float tsum = 0.f;
    float e0[8], e1[8];
#pragma unroll
    for (int r = 0; r < 8; ++r) {
      e0[r] = __expf(s0[r] - mn);
      e1[r] = __expf(s1[r] - mn);
      tsum += e0[r] + e1[r];
    }
    tsum += __shfl_xor(tsum, 16);
    l = l * corr + tsum;
    m = mn;
#pragma unroll
    for (int i = 0; i < 4; ++i)
#pragma unroll
      for (int r = 0; r < 8; ++r) acc[i][r] *= corr;

    // bounce P^T through per-wave LDS: Ps[q][key], key-contiguous
    half_t* P = Ps[wave] + ln15 * 32;
#pragma unroll
    for (int r = 0; r < 8; ++r) {
      P[r + 8 * hi]      = (half_t)e0[r];   // keys 0..15  (M = r + 8*hi)
      P[16 + r + 8 * hi] = (half_t)e1[r];   // keys 16..31
    }
    asm volatile("s_wait_dscnt 0x0" ::: "memory");  // cross-lane LDS handoff

    // out^T += V^T x P^T   (4 d-tiles of 16)
    v16h pb = load_fragB(Ps[wave] + ln15 * 32, hi);
#pragma unroll
    for (int dt = 0; dt < 4; ++dt) {
      v16h va = load_fragA(Vb + (dt * 16 + ln15) * 32, hi);
      acc[dt] = wmma_f16(va, pb, acc[dt]);
    }
  }

  const float inv = 1.f / l;
#pragma unroll
  for (int dt = 0; dt < 4; ++dt) {
#pragma unroll
    for (int r = 0; r < 8; ++r) {
      int d = dt * 16 + r + 8 * hi;
      O[((size_t)bi * SEQ + qg) * DIMD + h * DHEAD + d] = acc[dt][r] * inv;
    }
  }
}

// ---------------------------------------------------------------------------
extern "C" void kernel_launch(void* const* d_in, const int* in_sizes, int n_in,
                              void* d_out, int out_size, void* d_ws,
                              size_t ws_size, hipStream_t stream) {
  (void)in_sizes; (void)n_in; (void)out_size; (void)ws_size;
  const float* x     = (const float*)d_in[0];
  const float* Wqkv  = (const float*)d_in[1];
  const float* bqkv  = (const float*)d_in[2];
  const float* Wproj = (const float*)d_in[3];
  const float* bproj = (const float*)d_in[4];
  float* out = (float*)d_out;

  char* ws = (char*)d_ws;
  const size_t szHead = (size_t)BATCH * NHEADS * SEQ * DHEAD * sizeof(half_t);
  half_t* Qh = (half_t*)(ws);
  half_t* Kh = (half_t*)(ws + szHead);
  half_t* Vt = (half_t*)(ws + 2 * szHead);
  float*  Oa = (float*)(ws + 3 * szHead);   // attention out, fp32 [4096][768]

  dim3 g1(QKVCOL / 128, ROWS / 64);         // (18, 64)
  gemm_wmma<0><<<g1, 256, 0, stream>>>(x, Wqkv, bqkv, QKVCOL, Qh, Kh, Vt,
                                       nullptr);

  dim3 g2(SEQ / 64, BATCH * NHEADS);        // (32, 24)
  attn_kernel<<<g2, 128, 0, stream>>>(Qh, Kh, Vt, Oa);

  dim3 g3(DIMD / 128, ROWS / 64);           // (6, 64)
  gemm_wmma<1><<<g3, 256, 0, stream>>>(Oa, Wproj, bproj, DIMD, nullptr, nullptr,
                                       nullptr, out);
}
